// MultiHeadSelfAttention_12970801234468
// MI455X (gfx1250) — matrix operand order repaired and checked by reading, not compiled, not run on hardware
//
#include <hip/hip_runtime.h>
#include <hip/hip_bf16.h>

#define EMBED 1024
#define SEQ   2048
#define BATCH 2
#define NHEAD 16
#define HDIM  64
#define NTOK  (BATCH * SEQ) /* 4096 */

typedef __attribute__((ext_vector_type(16))) __bf16 v16bf;
typedef __attribute__((ext_vector_type(8)))  float  v8f;

union Frag16 { v16bf v; unsigned u[8]; };

__device__ __forceinline__ __bf16 f2bf(float f) {
  unsigned x = __builtin_bit_cast(unsigned, f);
  unsigned r = x + 0x7FFFu + ((x >> 16) & 1u);   // round-to-nearest-even
  return __builtin_bit_cast(__bf16, (unsigned short)(r >> 16));
}
__device__ __forceinline__ __bf16 to_bf(float f)  { return f2bf(f); }
__device__ __forceinline__ __bf16 to_bf(__bf16 f) { return f; }

__device__ __forceinline__ void store_c(float* p, float v)  { *p = v; }
__device__ __forceinline__ void store_c(__bf16* p, float v) { *p = f2bf(v); }

// 16-bit A-matrix VGPR k-offset (ISA 7.12.2): V0..3 -> K=2v+8*half, V4..7 -> 16+...
__device__ __forceinline__ int patA(int v, int half) {
  return ((v & 4) << 2) + ((v & 3) << 1) + (half << 3);
}
// 16-bit B-matrix VGPR k-offset (same as A): V0..3 -> K=2v+8*half, V4..7 -> 16+...
__device__ __forceinline__ int patB(int v, int half) {
  return ((v & 4) << 2) + ((v & 3) << 1) + (half << 3);
}

// --- 16-lane butterfly reductions via DPP16 ROW_XMASK (VALU, no LDS) -------
#if __has_builtin(__builtin_amdgcn_update_dpp)
template <int MASK>
__device__ __forceinline__ float dpp_xor_f(float x) {
  return __builtin_bit_cast(float,
      __builtin_amdgcn_update_dpp(0, __builtin_bit_cast(int, x),
                                  0x160 | MASK /*ROW_XMASK*/, 0xf, 0xf, true));
}
__device__ __forceinline__ float rmax16(float t) {
  t = fmaxf(t, dpp_xor_f<1>(t));
  t = fmaxf(t, dpp_xor_f<2>(t));
  t = fmaxf(t, dpp_xor_f<4>(t));
  t = fmaxf(t, dpp_xor_f<8>(t));
  return t;
}
__device__ __forceinline__ float rsum16(float t) {
  t += dpp_xor_f<1>(t);
  t += dpp_xor_f<2>(t);
  t += dpp_xor_f<4>(t);
  t += dpp_xor_f<8>(t);
  return t;
}
#else
__device__ __forceinline__ float rmax16(float t) {
  for (int off = 1; off < 16; off <<= 1) t = fmaxf(t, __shfl_xor(t, off, 32));
  return t;
}
__device__ __forceinline__ float rsum16(float t) {
  for (int off = 1; off < 16; off <<= 1) t += __shfl_xor(t, off, 32);
  return t;
}
#endif

// LDS byte offset of a __shared__ object (shared aperture: offset = addr[31:0])
#define LDS_OFF(p) ((unsigned)(size_t)(p))

__device__ __forceinline__ void async_b128(unsigned lds_off, const void* gaddr) {
  asm volatile("global_load_async_to_lds_b128 %0, %1, off"
               :: "v"(lds_off), "v"((unsigned long long)(size_t)gaddr)
               : "memory");
}
__device__ __forceinline__ void wait_async0() {
  asm volatile("s_wait_asynccnt 0x0" ::: "memory");
}

// ---------------------------------------------------------------------------
// C[M,N] = A[M,K] @ W[K,N] + bias  (bf16 WMMA, f32 accumulate)
// Block: 256 threads = 8 waves; block tile 128x128, BK=32.
// Wave tile 32x64 -> 2x4 grid of 16x16 WMMA accumulators.
// A tile staged via async-DMA when A is already bf16 in memory.
// ---------------------------------------------------------------------------
template <typename AT, typename CT>
__global__ __launch_bounds__(256) void gemm_bias_wmma(
    const AT* __restrict__ A, const float* __restrict__ W,
    const float* __restrict__ bias, CT* __restrict__ C,
    int M, int N, int K)
{
  constexpr bool kAsyncA = __is_same(AT, __bf16);

  __shared__ __bf16 As[128][40];   // [row][k], 80B row stride (16B aligned)
  __shared__ __bf16 Bs[128][34];   // [col][k] (W tile transposed; pairs contiguous)

  const int tid  = threadIdx.x;
  const int lane = tid & 31;
  const int wid  = tid >> 5;      // 0..7
  const int wm   = wid & 3;       // 4 row groups of 32
  const int wn   = wid >> 2;      // 2 col groups of 64
  const int half = lane >> 4;
  const int l16  = lane & 15;

  const int rowBase = blockIdx.y * 128;
  const int colBase = blockIdx.x * 128;

  v8f acc[2][4] = {};

  for (int kt = 0; kt < K; kt += 32) {
    // ---- stage A tile 128x32 bf16 ----
    if constexpr (kAsyncA) {
      // 512 16B segments, 2 per thread, DMA straight into LDS
      #pragma unroll
      for (int i = 0; i < 2; ++i) {
        int q = tid + i * 256;
        int r = q >> 2, s = q & 3;             // 4 segs per 64B row
        async_b128(LDS_OFF(&As[r][s * 8]),
                   A + (size_t)(rowBase + r) * K + kt + s * 8);
      }
    } else {
      int r  = tid >> 1;
      int c0 = (tid & 1) << 4;
      const AT* src = A + (size_t)(rowBase + r) * K + kt + c0;
      #pragma unroll
      for (int i = 0; i < 16; ++i) As[r][c0 + i] = to_bf(src[i]);
    }
    // ---- stage W tile 32x128, transposed into Bs[col][k] ----
    {
      int r  = tid >> 3;            // k 0..31
      int c0 = (tid & 7) << 4;      // col 0..112
      const float* src = W + (size_t)(kt + r) * N + colBase + c0;
      #pragma unroll
      for (int i = 0; i < 16; ++i) Bs[c0 + i][r] = f2bf(src[i]);
    }
    if constexpr (kAsyncA) wait_async0();
    __syncthreads();

    Frag16 a[2], b[4];
    #pragma unroll
    for (int mi = 0; mi < 2; ++mi) {
      int row = wm * 32 + mi * 16 + l16;
      #pragma unroll
      for (int v = 0; v < 8; ++v)
        a[mi].u[v] = *(const unsigned*)&As[row][patA(v, half)];
    }
    #pragma unroll
    for (int ni = 0; ni < 4; ++ni) {
      int col = wn * 64 + ni * 16 + l16;
      #pragma unroll
      for (int v = 0; v < 8; ++v)
        b[ni].u[v] = *(const unsigned*)&Bs[col][patB(v, half)];
    }
    #pragma unroll
    for (int mi = 0; mi < 2; ++mi)
      #pragma unroll
      for (int ni = 0; ni < 4; ++ni)
        acc[mi][ni] = __builtin_amdgcn_wmma_f32_16x16x32_bf16(
            false, a[mi].v, false, b[ni].v, (short)0, acc[mi][ni], false, false);
    __syncthreads();
  }

  // store: C/D layout row M = v + 8*half, col N = l16
  #pragma unroll
  for (int ni = 0; ni < 4; ++ni) {
    int gcol = colBase + wn * 64 + ni * 16 + l16;
    float bv = bias[gcol];
    #pragma unroll
    for (int mi = 0; mi < 2; ++mi)
      #pragma unroll
      for (int v = 0; v < 8; ++v) {
        int grow = rowBase + wm * 32 + mi * 16 + v + 8 * half;
        store_c(&C[(size_t)grow * N + gcol], acc[mi][ni][v] + bv);
      }
  }
}

// ---------------------------------------------------------------------------
// Flash attention: block = 4 waves (64 query rows of one (b,h)); keys are
// streamed in 32-row blocks, K/V tiles DMA'd into LDS once per block
// (double-buffered, async), shared by all 4 waves. Logits never touch HBM.
// ---------------------------------------------------------------------------
__global__ __launch_bounds__(128) void attn_wmma(
    const __bf16* __restrict__ Qb, const __bf16* __restrict__ Kb,
    const __bf16* __restrict__ Vb, __bf16* __restrict__ Ab)
{
  __shared__ __bf16 Ks[2][32][72];  // 144B row stride (16B aligned)
  __shared__ __bf16 Vs[2][32][72];
  __shared__ float  Pt[4][16][33];  // per-wave P staging (C-layout -> A-layout)

  const int tid  = threadIdx.x;
  const int w    = tid >> 5;
  const int lane = tid & 31;
  const int half = lane >> 4;
  const int l16  = lane & 15;
  const int b    = blockIdx.z;
  const int h    = blockIdx.y;
  const int q0   = blockIdx.x * 64 + w * 16;

  const size_t batchOff = (size_t)b * SEQ * EMBED;
  const size_t headOff  = (size_t)h * HDIM;
  const __bf16* Qp = Qb + batchOff + headOff;
  const __bf16* Kp = Kb + batchOff + headOff;
  const __bf16* Vp = Vb + batchOff + headOff;

  // DMA one 32x64 K tile + V tile into LDS buffer `buf` (4 ops/thread)
  auto issue_tiles = [&](int buf, int jrow) {
    #pragma unroll
    for (int i = 0; i < 2; ++i) {
      int q = tid + i * 128;          // 256 segments of 16B per tile
      int r = q >> 3, s = q & 7;      // 8 segs per 128B row
      const size_t goff = (size_t)(jrow + r) * EMBED + s * 8;
      async_b128(LDS_OFF(&Ks[buf][r][s * 8]), Kp + goff);
      async_b128(LDS_OFF(&Vs[buf][r][s * 8]), Vp + goff);
    }
  };

  // Q tile 16x64 as two 16x32 A fragments (resident for whole kernel)
  Frag16 aq[2];
  {
    const __bf16* qrow = Qp + (size_t)(q0 + l16) * EMBED;
    #pragma unroll
    for (int c = 0; c < 2; ++c)
      #pragma unroll
      for (int v = 0; v < 8; ++v)
        aq[c].u[v] = *(const unsigned*)(qrow + c * 32 + patA(v, half));
  }

  v8f o[4] = {};                 // O accumulator 16x64 (C layout)
  float m_r[8], l_r[8];          // per-row running max / sum (rows v+8*half)
  #pragma unroll
  for (int v = 0; v < 8; ++v) { m_r[v] = -3.0e38f; l_r[v] = 0.0f; }

  issue_tiles(0, 0);             // prime the pipeline

  for (int j = 0; j < SEQ; j += 32) {
    const int cur = (j >> 5) & 1;
    wait_async0();               // this wave's DMA for `cur` complete
    __syncthreads();             // all waves' DMA visible; prev readers done
    if (j + 32 < SEQ) issue_tiles(cur ^ 1, j + 32);  // overlap next DMA

    // ---- S = Q @ K^T for 32 keys: two 16x16 logit tiles ----
    // Load ALL K fragments into distinct registers first so the four WMMAs
    // overlap the remaining LDS loads (no full s_wait_dscnt per WMMA).
    Frag16 bk[2][2];
    #pragma unroll
    for (int jj = 0; jj < 2; ++jj) {
      const __bf16* krow = &Ks[cur][jj * 16 + l16][0];
      #pragma unroll
      for (int c = 0; c < 2; ++c)
        #pragma unroll
        for (int v = 0; v < 8; ++v)
          bk[jj][c].u[v] = *(const unsigned*)(krow + c * 32 + patB(v, half));
    }
    v8f s[2] = {};
    #pragma unroll
    for (int jj = 0; jj < 2; ++jj)
      #pragma unroll
      for (int c = 0; c < 2; ++c)
        s[jj] = __builtin_amdgcn_wmma_f32_16x16x32_bf16(
            false, aq[c].v, false, bk[jj][c].v, (short)0, s[jj], false, false);

    // ---- online softmax (rows live in 16-lane halves of each VGPR) ----
    float alpha[8];
    #pragma unroll
    for (int v = 0; v < 8; ++v) {
      float x0 = s[0][v] * 0.125f;     // 1/sqrt(64)
      float x1 = s[1][v] * 0.125f;
      s[0][v] = x0; s[1][v] = x1;
      float t  = rmax16(fmaxf(x0, x1));
      float nm = fmaxf(m_r[v], t);
      alpha[v] = __expf(m_r[v] - nm);
      m_r[v] = nm;
    }
    #pragma unroll
    for (int v = 0; v < 8; ++v) {
      float p0 = __expf(s[0][v] - m_r[v]);
      float p1 = __expf(s[1][v] - m_r[v]);
      s[0][v] = p0; s[1][v] = p1;
      l_r[v] = l_r[v] * alpha[v] + rsum16(p0 + p1);
    }
    #pragma unroll
    for (int n = 0; n < 4; ++n)
      #pragma unroll
      for (int v = 0; v < 8; ++v)
        o[n][v] *= alpha[v];         // C layout rows match m/l rows: no shuffle

    // ---- P: C-layout -> A-layout via per-wave LDS tile ----
    #pragma unroll
    for (int v = 0; v < 8; ++v) {
      int r = v + 8 * half;
      Pt[w][r][l16]      = s[0][v];
      Pt[w][r][16 + l16] = s[1][v];
    }
    __syncthreads();
    Frag16 ap;
    #pragma unroll
    for (int v = 0; v < 8; ++v) {
      int k0 = patA(v, half);
      __bf16 lo = f2bf(Pt[w][l16][k0]);
      __bf16 hi = f2bf(Pt[w][l16][k0 + 1]);
      ap.u[v] = ((unsigned)__builtin_bit_cast(unsigned short, hi) << 16)
              |  (unsigned)__builtin_bit_cast(unsigned short, lo);
    }

    // ---- O += P(16x32) @ V(32x64), V from LDS ----
    // Same pattern: gather all four V fragments, then issue the four WMMAs.
    Frag16 bvf[4];
    #pragma unroll
    for (int n = 0; n < 4; ++n)
      #pragma unroll
      for (int v = 0; v < 8; ++v) {
        int kk = patB(v, half);
        const __bf16* vp = &Vs[cur][kk][n * 16 + l16];
        __bf16 lo = vp[0];
        __bf16 hi = vp[72];          // next key row (stride 72 elems)
        bvf[n].u[v] = ((unsigned)__builtin_bit_cast(unsigned short, hi) << 16)
                    |  (unsigned)__builtin_bit_cast(unsigned short, lo);
      }
    #pragma unroll
    for (int n = 0; n < 4; ++n)
      o[n] = __builtin_amdgcn_wmma_f32_16x16x32_bf16(
          false, ap.v, false, bvf[n].v, (short)0, o[n], false, false);
  }

  // ---- normalize + write concat layout [B,S,E] (head h at cols h*64..) ----
  #pragma unroll
  for (int v = 0; v < 8; ++v) l_r[v] = 1.0f / l_r[v];
  __bf16* Aout = Ab + batchOff + headOff;
  #pragma unroll
  for (int n = 0; n < 4; ++n)
    #pragma unroll
    for (int v = 0; v < 8; ++v) {
      int r = q0 + v + 8 * half;
      Aout[(size_t)r * EMBED + n * 16 + l16] = f2bf(o[n][v] * l_r[v]);
    }
}

// ---------------------------------------------------------------------------
extern "C" void kernel_launch(void* const* d_in, const int* in_sizes, int n_in,
                              void* d_out, int out_size, void* d_ws, size_t ws_size,
                              hipStream_t stream)
{
  (void)in_sizes; (void)n_in; (void)out_size; (void)ws_size;
  const float* q  = (const float*)d_in[0];
  const float* k  = (const float*)d_in[1];
  const float* v  = (const float*)d_in[2];
  const float* Wq = (const float*)d_in[3];
  const float* bq = (const float*)d_in[4];
  const float* Wk = (const float*)d_in[5];
  const float* bk = (const float*)d_in[6];
  const float* Wv = (const float*)d_in[7];
  const float* bv = (const float*)d_in[8];
  const float* Wo = (const float*)d_in[9];
  const float* bo = (const float*)d_in[10];
  float* out = (float*)d_out;

  // workspace: 4 bf16 buffers of NTOK*EMBED = 33.5 MB total
  __bf16* Qb = (__bf16*)d_ws;
  __bf16* Kb = Qb + (size_t)NTOK * EMBED;
  __bf16* Vb = Kb + (size_t)NTOK * EMBED;
  __bf16* Ab = Vb + (size_t)NTOK * EMBED;

  dim3 gblock(256);
  dim3 ggrid(EMBED / 128, NTOK / 128);   // (8, 32)
  gemm_bias_wmma<float, __bf16><<<ggrid, gblock, 0, stream>>>(q, Wq, bq, Qb, NTOK, EMBED, EMBED);
  gemm_bias_wmma<float, __bf16><<<ggrid, gblock, 0, stream>>>(k, Wk, bk, Kb, NTOK, EMBED, EMBED);
  gemm_bias_wmma<float, __bf16><<<ggrid, gblock, 0, stream>>>(v, Wv, bv, Vb, NTOK, EMBED, EMBED);

  dim3 ablock(128);
  dim3 agrid(SEQ / 64, NHEAD, BATCH);    // (32, 16, 2)
  attn_wmma<<<agrid, ablock, 0, stream>>>(Qb, Kb, Vb, Ab);

  gemm_bias_wmma<__bf16, float><<<ggrid, gblock, 0, stream>>>(Ab, Wo, bo, out, NTOK, EMBED, EMBED);
}
